// GraphormerLayer_74320114090196
// MI455X (gfx1250) — matrix operand order repaired and checked by reading, not compiled, not run on hardware
//
#include <hip/hip_runtime.h>
#include <hip/hip_bf16.h>
#include <stdint.h>

#define BB 4
#define NN 2048
#define EE 256
#define HH 8
#define DD 32
#define MM (BB*NN)   // 8192 rows

typedef __attribute__((ext_vector_type(16))) __bf16 v16bf;
typedef __attribute__((ext_vector_type(8)))  __bf16 v8bf;
typedef __attribute__((ext_vector_type(8)))  float  v8f;

__device__ __forceinline__ v16bf load_frag(const __bf16* p0, const __bf16* p1) {
    v8bf a = *(const v8bf*)p0;
    v8bf b = *(const v8bf*)p1;
    v16bf r;
#pragma unroll
    for (int i = 0; i < 8; ++i) { r[i] = a[i]; r[i+8] = b[i]; }
    return r;
}

__device__ __forceinline__ v8f wmma_bf16(v16bf a, v16bf b, v8f c) {
    return __builtin_amdgcn_wmma_f32_16x16x32_bf16(false, a, false, b, (short)0, c,
                                                   false, false);
}

// async global -> LDS copy of one 16B chunk (CDNA5, tracked by ASYNCcnt)
__device__ __forceinline__ void async_ld16(uint32_t lds_off, const void* g) {
    asm volatile("global_load_async_to_lds_b128 %0, %1, off"
                 :: "v"(lds_off), "v"((uint64_t)(uintptr_t)g) : "memory");
}
__device__ __forceinline__ void wait_async0() {
    asm volatile("s_wait_asynccnt 0" ::: "memory");
}

// ---- weight transpose + fp32->bf16 convert: WT[n][k] = W[k][n] -------------
__global__ void transpose_w(const float* __restrict__ W, __bf16* __restrict__ WT) {
    int idx = blockIdx.x * 256 + threadIdx.x;   // idx = k*E + n
    int k = idx >> 8, n = idx & 255;
    WT[(size_t)n * EE + k] = (__bf16)W[idx];
}

// ---- LayerNorm rows of 256, one wave per row, bf16 out ---------------------
__global__ void layernorm_bf16(const float* __restrict__ x, const float* __restrict__ g,
                               const float* __restrict__ be, __bf16* __restrict__ out) {
    int wv = threadIdx.x >> 5, lane = threadIdx.x & 31;
    size_t row = (size_t)blockIdx.x * 8 + wv;
    const float* xr = x + row * EE;
    float v[8], s = 0.f, s2 = 0.f;
#pragma unroll
    for (int j = 0; j < 8; ++j) { v[j] = xr[lane + 32*j]; s += v[j]; s2 += v[j]*v[j]; }
#pragma unroll
    for (int off = 1; off < 32; off <<= 1) {
        s  += __shfl_xor(s,  off, 32);
        s2 += __shfl_xor(s2, off, 32);
    }
    float mu  = s * (1.0f/EE);
    float var = s2 * (1.0f/EE) - mu*mu;
    float rs  = rsqrtf(var + 1e-5f);
#pragma unroll
    for (int j = 0; j < 8; ++j) {
        int e = lane + 32*j;
        out[row*EE + e] = (__bf16)((v[j]-mu)*rs*g[e] + be[e]);
    }
}

// ---- generic 256-K bf16 WMMA GEMM, one 16x16 C tile per wave ---------------
// MODE 0: out bf16 [b,h,tok,d]          (Q, K)
// MODE 1: out bf16 [b,h,d,tok]          (V transposed)
// MODE 2: out f32  = C + bias + resid   (O-proj -> h, FFN2 -> final)
// MODE 3: out bf16 = relu(C + bias)     (FFN1)
template <int MODE>
__global__ __launch_bounds__(128) void gemm256(const __bf16* __restrict__ A,
                                               const __bf16* __restrict__ BT,
                                               const float* __restrict__ bias,
                                               const float* __restrict__ resid,
                                               void* __restrict__ outp) {
    int wv = threadIdx.x >> 5, lane = threadIdx.x & 31;
    int hl = lane >> 4, nl = lane & 15;
    int m0 = (blockIdx.y * 4 + wv) * 16;
    int n0 = blockIdx.x * 16;
    v8f acc = {0.f,0.f,0.f,0.f,0.f,0.f,0.f,0.f};
#pragma unroll
    for (int k0 = 0; k0 < EE; k0 += 32) {
        const __bf16* ap = A  + (size_t)(m0 + nl) * EE + k0 + hl*8;
        const __bf16* bp = BT + (size_t)(n0 + nl) * EE + k0 + hl*8;
        v16bf af  = load_frag(ap, ap + 16);
        v16bf bfr = load_frag(bp, bp + 16);
        acc = wmma_bf16(af, bfr, acc);
    }
#pragma unroll
    for (int r = 0; r < 8; ++r) {
        int m = m0 + r + hl*8;
        int n = n0 + nl;
        if (MODE == 0) {
            int b = m / NN, tok = m % NN, h = n / DD, d = n % DD;
            ((__bf16*)outp)[(((size_t)b*HH + h)*NN + tok)*DD + d] = (__bf16)acc[r];
        } else if (MODE == 1) {
            int b = m / NN, tok = m % NN, h = n / DD, d = n % DD;
            ((__bf16*)outp)[(((size_t)b*HH + h)*DD + d)*NN + tok] = (__bf16)acc[r];
        } else if (MODE == 2) {
            ((float*)outp)[(size_t)m*EE + n] =
                acc[r] + bias[n] + resid[(size_t)m*EE + n];
        } else {
            ((__bf16*)outp)[(size_t)m*EE + n] = (__bf16)fmaxf(acc[r] + bias[n], 0.f);
        }
    }
}

// ---- streaming (flash) attention ------------------------------------------
// One wave handles 16 queries of one head; the 4 waves of a block share one
// (b,h), so the 32-key K/V tiles are staged once per block into LDS with
// double-buffered GLOBAL_LOAD_ASYNC_TO_LDS_B128 (ASYNCcnt) copies.
// S = QK^T/sqrt(D) + iw1*infl + ib1 ; online softmax ; (iw2*infl+ib2) folded
// into the numerator P (the post-softmax multiply is linear in k).
__global__ __launch_bounds__(128) void attn_flash(const __bf16* __restrict__ Q,
                                                  const __bf16* __restrict__ K,
                                                  const __bf16* __restrict__ Vt,
                                                  const float* __restrict__ infl,
                                                  const float* __restrict__ iw1p,
                                                  const float* __restrict__ ib1p,
                                                  const float* __restrict__ iw2p,
                                                  const float* __restrict__ ib2p,
                                                  __bf16* __restrict__ Out) {
    __shared__ __bf16 kvbuf[2][2048];    // [buf][ K tile 32x32 | V tile 32x32 ]
    __shared__ __bf16 plds[4][16*32];    // per-wave P relayout tile
    int tid = threadIdx.x;
    int wv = tid >> 5, lane = tid & 31;
    int hl = lane >> 4, nl = lane & 15;
    int gw = blockIdx.x * 4 + wv;          // global wave id over B*H*(N/16)
    int qt = gw & (NN/16 - 1);
    int bh = gw >> 7;                      // N/16 = 128; uniform per block (128%4==0)
    int b  = bh / HH, h = bh % HH;
    int q0 = qt * 16;
    const float iw1 = iw1p[0], ib1 = ib1p[0], iw2 = iw2p[0], ib2 = ib2p[0];
    const float scale = 0.17677669529663687f;   // 1/sqrt(32)

    const __bf16* qbase = Q + ((size_t)bh*NN + q0 + nl) * DD + hl*8;
    v16bf qf = load_frag(qbase, qbase + 16);

    v8f acc0 = {0.f,0.f,0.f,0.f,0.f,0.f,0.f,0.f};
    v8f acc1 = {0.f,0.f,0.f,0.f,0.f,0.f,0.f,0.f};
    float mrun[8], lrun[8];
#pragma unroll
    for (int r = 0; r < 8; ++r) { mrun[r] = -3.0e38f; lrun[r] = 0.f; }

    const float* irow = infl + (size_t)b * NN * NN;
    __bf16* pl = plds[wv];

    // --- per-thread staging chunk: row ci (key for K / d for V), 16B part cp
    int ci = tid >> 2, cp = tid & 3;
    const __bf16* kg0 = K  + ((size_t)bh*NN + ci)*DD + cp*8;  // advance kt*DD
    const __bf16* vg0 = Vt + ((size_t)bh*DD + ci)*NN + cp*8;  // advance kt
    uint32_t koff0 = (uint32_t)(uintptr_t)&kvbuf[0][ci*32 + cp*8];
    uint32_t voff0 = (uint32_t)(uintptr_t)&kvbuf[0][1024 + ci*32 + cp*8];
    uint32_t koff1 = (uint32_t)(uintptr_t)&kvbuf[1][ci*32 + cp*8];
    uint32_t voff1 = (uint32_t)(uintptr_t)&kvbuf[1][1024 + ci*32 + cp*8];

    // prefetch tile 0 into buffer 0
    async_ld16(koff0, kg0);
    async_ld16(voff0, vg0);

    for (int it = 0; it < NN/32; ++it) {
        int kt = it * 32;
        int cur = it & 1;
        wait_async0();          // my async copies for buf[cur] are done
        __syncthreads();        // everyone's copies done; prev reads of buf[!cur] done
        if (it + 1 < NN/32) {   // double-buffer: start tile it+1 while computing it
            async_ld16(cur ? koff0 : koff1, kg0 + (size_t)(kt + 32) * DD);
            async_ld16(cur ? voff0 : voff1, vg0 + (kt + 32));
            // warm L2 for the next influence-matrix span (global_prefetch_b8)
            __builtin_prefetch(&irow[(size_t)(q0 + nl)*NN + kt + 32 + hl*16], 0, 0);
        }
        const __bf16* kb = &kvbuf[cur][0];
        const __bf16* vb = &kvbuf[cur][1024];

        // K^T B-fragments from LDS (key rows are contiguous 32-half rows)
        const __bf16* kb0 = kb + nl*32 + hl*8;
        const __bf16* kb1 = kb + (16 + nl)*32 + hl*8;
        v16bf bk0 = load_frag(kb0, kb0 + 16);
        v16bf bk1 = load_frag(kb1, kb1 + 16);
        v8f s0 = {0.f,0.f,0.f,0.f,0.f,0.f,0.f,0.f};
        v8f s1 = {0.f,0.f,0.f,0.f,0.f,0.f,0.f,0.f};
        s0 = wmma_bf16(qf, bk0, s0);
        s1 = wmma_bf16(qf, bk1, s1);

        float i0[8], i1[8];
#pragma unroll
        for (int r = 0; r < 8; ++r) {
            int q = q0 + r + hl*8;
            i0[r] = irow[(size_t)q*NN + kt + nl];
            i1[r] = irow[(size_t)q*NN + kt + 16 + nl];
            s0[r] = s0[r]*scale + iw1*i0[r] + ib1;
            s1[r] = s1[r]*scale + iw1*i1[r] + ib1;
        }
#pragma unroll
        for (int r = 0; r < 8; ++r) {
            float mx = fmaxf(s0[r], s1[r]);
#pragma unroll
            for (int off = 1; off < 16; off <<= 1) mx = fmaxf(mx, __shfl_xor(mx, off, 32));
            float mnew = fmaxf(mrun[r], mx);
            float corr = __expf(mrun[r] - mnew);
            float p0 = __expf(s0[r] - mnew);
            float p1 = __expf(s1[r] - mnew);
            float rs = p0 + p1;
#pragma unroll
            for (int off = 1; off < 16; off <<= 1) rs += __shfl_xor(rs, off, 32);
            lrun[r] = lrun[r]*corr + rs;
            mrun[r] = mnew;
            acc0[r] *= corr; acc1[r] *= corr;
            s0[r] = p0 * (iw2*i0[r] + ib2);   // post-softmax influence weight
            s1[r] = p1 * (iw2*i1[r] + ib2);
        }
        // relayout P (C-layout f32) -> A-fragment via per-wave LDS tile
#pragma unroll
        for (int r = 0; r < 8; ++r) {
            pl[(r + hl*8)*32 + nl]      = (__bf16)s0[r];
            pl[(r + hl*8)*32 + 16 + nl] = (__bf16)s1[r];
        }
        v16bf pf = load_frag(pl + nl*32 + hl*8, pl + nl*32 + hl*8 + 16);
        // V B-fragments from LDS (d rows are contiguous 32-half rows)
        const __bf16* v0 = vb + nl*32 + hl*8;
        const __bf16* v1 = vb + (16 + nl)*32 + hl*8;
        v16bf bv0 = load_frag(v0, v0 + 16);
        v16bf bv1 = load_frag(v1, v1 + 16);
        acc0 = wmma_bf16(pf, bv0, acc0);
        acc1 = wmma_bf16(pf, bv1, acc1);
    }
#pragma unroll
    for (int r = 0; r < 8; ++r) {
        int q = q0 + r + hl*8;
        float inv = 1.0f / lrun[r];
        size_t row = ((size_t)b*NN + q)*EE + h*DD;
        Out[row + nl]      = (__bf16)(acc0[r]*inv);
        Out[row + 16 + nl] = (__bf16)(acc1[r]*inv);
    }
}

extern "C" void kernel_launch(void* const* d_in, const int* in_sizes, int n_in,
                              void* d_out, int out_size, void* d_ws, size_t ws_size,
                              hipStream_t stream) {
    const float* x    = (const float*)d_in[0];
    const float* infl = (const float*)d_in[1];
    const float* Wq   = (const float*)d_in[2];
    const float* Wk   = (const float*)d_in[3];
    const float* Wv   = (const float*)d_in[4];
    const float* Wo   = (const float*)d_in[5];
    const float* bo   = (const float*)d_in[6];
    const float* iw1  = (const float*)d_in[7];
    const float* ib1  = (const float*)d_in[8];
    const float* iw2  = (const float*)d_in[9];
    const float* ib2  = (const float*)d_in[10];
    const float* W1   = (const float*)d_in[11];
    const float* b1   = (const float*)d_in[12];
    const float* W2   = (const float*)d_in[13];
    const float* b2   = (const float*)d_in[14];
    const float* g1   = (const float*)d_in[15];
    const float* be1  = (const float*)d_in[16];
    const float* g2   = (const float*)d_in[17];
    const float* be2  = (const float*)d_in[18];

    char* wsp = (char*)d_ws;
    auto alloc = [&](size_t bytes) -> char* {
        char* p = wsp; wsp += (bytes + 255) & ~(size_t)255; return p;
    };
    const size_t act_bf16 = (size_t)MM * EE * sizeof(__bf16);     // 4 MB
    __bf16* xn   = (__bf16*)alloc(act_bf16);
    __bf16* Qb   = (__bf16*)alloc(act_bf16);
    __bf16* Kb   = (__bf16*)alloc(act_bf16);
    __bf16* Vtb  = (__bf16*)alloc(act_bf16);
    __bf16* attn = (__bf16*)alloc(act_bf16);
    float*  hbuf = (float*) alloc((size_t)MM * EE * sizeof(float)); // 8 MB
    __bf16* hn   = (__bf16*)alloc(act_bf16);
    __bf16* f1   = (__bf16*)alloc(act_bf16);
    const size_t wbytes = (size_t)EE * EE * sizeof(__bf16);
    __bf16* WqT = (__bf16*)alloc(wbytes);
    __bf16* WkT = (__bf16*)alloc(wbytes);
    __bf16* WvT = (__bf16*)alloc(wbytes);
    __bf16* WoT = (__bf16*)alloc(wbytes);
    __bf16* W1T = (__bf16*)alloc(wbytes);
    __bf16* W2T = (__bf16*)alloc(wbytes);

    transpose_w<<<256, 256, 0, stream>>>(Wq, WqT);
    transpose_w<<<256, 256, 0, stream>>>(Wk, WkT);
    transpose_w<<<256, 256, 0, stream>>>(Wv, WvT);
    transpose_w<<<256, 256, 0, stream>>>(Wo, WoT);
    transpose_w<<<256, 256, 0, stream>>>(W1, W1T);
    transpose_w<<<256, 256, 0, stream>>>(W2, W2T);

    // LN1
    layernorm_bf16<<<MM/8, 256, 0, stream>>>(x, g1, be1, xn);

    dim3 ggrid(EE/16, MM/64);   // 16 x 128, 4 waves per block
    // Q, K: [b,h,tok,d] ; V: transposed [b,h,d,tok]
    gemm256<0><<<ggrid, 128, 0, stream>>>(xn, WqT, nullptr, nullptr, Qb);
    gemm256<0><<<ggrid, 128, 0, stream>>>(xn, WkT, nullptr, nullptr, Kb);
    gemm256<1><<<ggrid, 128, 0, stream>>>(xn, WvT, nullptr, nullptr, Vtb);

    // flash attention: B*H*(N/16) = 4096 waves, 4 per block
    attn_flash<<<1024, 128, 0, stream>>>(Qb, Kb, Vtb, infl, iw1, ib1, iw2, ib2, attn);

    // h = attn @ Wo + bo + x
    gemm256<2><<<ggrid, 128, 0, stream>>>(attn, WoT, bo, x, hbuf);
    // LN2
    layernorm_bf16<<<MM/8, 256, 0, stream>>>(hbuf, g2, be2, hn);
    // FFN
    gemm256<3><<<ggrid, 128, 0, stream>>>(hn, W1T, b1, nullptr, f1);
    gemm256<2><<<ggrid, 128, 0, stream>>>(f1, W2T, b2, hbuf, (float*)d_out);
}
